// CrossAttention_44032004719071
// MI455X (gfx1250) — matrix operand order repaired and checked by reading, not compiled, not run on hardware
//
#include <hip/hip_runtime.h>

#define DMODEL 1024
#define NHEADS 16
#define DH     64
#define SEQ    2048
#define BATCH  2
#define MROWS  (BATCH * SEQ)   // 4096
#define LN_EPS 1e-5f

typedef _Float16 half_t;
typedef __attribute__((ext_vector_type(16))) _Float16 v16h;
typedef __attribute__((ext_vector_type(8)))  _Float16 v8h;
typedef __attribute__((ext_vector_type(4)))  _Float16 v4h;
typedef __attribute__((ext_vector_type(8)))  float    v8f;
typedef int v4i __attribute__((vector_size(16)));   // matches builtin's param type

#if __has_builtin(__builtin_amdgcn_global_load_async_to_lds_b128)
#define HAVE_ASYNC_LDS 1
#else
#define HAVE_ASYNC_LDS 0
#endif

// ---------------------------------------------------------------------------
// Async global->LDS copy of one 16-byte chunk (CDNA5 ASYNCcnt path), with a
// guaranteed-compilable synchronous fallback (load -> ds_store).
// Builtin signature (from hipcc diagnostic): param0 = v4i __device__* (AS1),
// param1 = LDS pointer, then imm offset + imm cpol.
// ---------------------------------------------------------------------------
static __device__ inline void async_copy16(const half_t* g, half_t* l) {
#if HAVE_ASYNC_LDS
  __builtin_amdgcn_global_load_async_to_lds_b128(
      (__attribute__((address_space(1))) v4i*)g,
      (__attribute__((address_space(3))) v4i*)l, 0, 0);
#else
  *(v8h*)l = *(const v8h*)g;
#endif
}

static __device__ inline void wait_async() {
#if HAVE_ASYNC_LDS
#if __has_builtin(__builtin_amdgcn_s_wait_asynccnt)
  __builtin_amdgcn_s_wait_asynccnt(0);
#else
  asm volatile("s_wait_asynccnt 0x0" ::: "memory");
#endif
#endif
}

// ---------------------------------------------------------------------------
// Fragment loaders (wave32 WMMA 16x16x32 f16; this code's contiguous k order, same for A and B)
// A (16x32, row-major source): lane m (0-15) holds row m, K = k0+0..15 contiguous;
//                              lane m+16 holds row m, K = k0+16..31.
// B (32x16) loaded from a K-contiguous (transposed) source: lane n (0-15) holds
//   col n, K = k0+0..15 contiguous; lane n+16 holds col n, K = k0+16..31.
// Work for both global and LDS base pointers (addrspace inferred after inline).
// ---------------------------------------------------------------------------
static __device__ inline v16h load_frag_a(const half_t* base, int ld, int row, int k0) {
  const int lane = threadIdx.x & 31;
  const int m  = lane & 15;
  const int hi = lane >> 4;
  const half_t* p = base + (size_t)(row + m) * ld + k0 + hi * 16;
  v8h lo = *(const v8h*)p;
  v8h hs = *(const v8h*)(p + 8);
  v16h r;
#pragma unroll
  for (int i = 0; i < 8; ++i) { r[i] = lo[i]; r[8 + i] = hs[i]; }
  return r;
}

static __device__ inline v16h load_frag_bt(const half_t* base, int ld, int col, int k0) {
  const int lane = threadIdx.x & 31;
  const int n  = lane & 15;
  const int hi = lane >> 4;
  const half_t* p = base + (size_t)(col + n) * ld + k0 + hi * 16;
  v8h lo = *(const v8h*)p;
  v8h hs = *(const v8h*)(p + 8);
  v16h r;
#pragma unroll
  for (int i = 0; i < 8; ++i) { r[i] = lo[i]; r[8 + i] = hs[i]; }
  return r;
}

// ---------------------------------------------------------------------------
// fp32 -> fp16 conversion, 4 elements/thread
// ---------------------------------------------------------------------------
__global__ __launch_bounds__(256) void cvt_f32_to_f16(
    const float* __restrict__ src, half_t* __restrict__ dst, int n4) {
  int i = blockIdx.x * blockDim.x + threadIdx.x;
  if (i >= n4) return;
  const float4 v = ((const float4*)src)[i];
  v4h o;
  o[0] = (half_t)v.x; o[1] = (half_t)v.y; o[2] = (half_t)v.z; o[3] = (half_t)v.w;
  ((v4h*)dst)[i] = o;
}

// ---------------------------------------------------------------------------
// Projection GEMM: Y[m][o] = sum_i X[m][i] * W[o][i] + bias[o]
// Wave tile 32(M) x 64(N); WG = 8 waves (4 along M, 2 along N) => 128x128.
// Output stored f16: storeVT=0 -> [B][H][S][dh]; storeVT=1 -> [B][H][dh][S].
// ---------------------------------------------------------------------------
__global__ __launch_bounds__(256) void proj_gemm(
    const half_t* __restrict__ X, const half_t* __restrict__ W,
    const float* __restrict__ bias, half_t* __restrict__ dst, int storeVT) {
  const int wave = threadIdx.x >> 5;
  const int lane = threadIdx.x & 31;
  const int wm = wave & 3, wn = wave >> 2;
  const int m0 = blockIdx.x * 128 + wm * 32;
  const int n0 = blockIdx.y * 128 + wn * 64;

  v8f acc[2][4] = {};

  for (int k0 = 0; k0 < DMODEL; k0 += 32) {
    v16h a0 = load_frag_a(X, DMODEL, m0,      k0);
    v16h a1 = load_frag_a(X, DMODEL, m0 + 16, k0);
    v16h bfr[4];
#pragma unroll
    for (int j = 0; j < 4; ++j) bfr[j] = load_frag_bt(W, DMODEL, n0 + j * 16, k0);
#pragma unroll
    for (int j = 0; j < 4; ++j) {
      acc[0][j] = __builtin_amdgcn_wmma_f32_16x16x32_f16(
          false, a0, false, bfr[j], (short)0, acc[0][j], false, false);
      acc[1][j] = __builtin_amdgcn_wmma_f32_16x16x32_f16(
          false, a1, false, bfr[j], (short)0, acc[1][j], false, false);
    }
  }

  const int hi = lane >> 4;
#pragma unroll
  for (int j = 0; j < 4; ++j) {
    const int col = n0 + j * 16 + (lane & 15);
    const float bv = bias[col];
    const int h = col >> 6, c = col & 63;
#pragma unroll
    for (int i = 0; i < 2; ++i) {
#pragma unroll
      for (int r = 0; r < 8; ++r) {
        const int mr = m0 + i * 16 + hi * 8 + r;
        const int bb = mr >> 11, s = mr & (SEQ - 1);
        const half_t hv = (half_t)(acc[i][j][r] + bv);
        const size_t idx = storeVT
            ? ((size_t)((bb * NHEADS + h) * DH + c) * SEQ + s)
            : ((size_t)((bb * NHEADS + h) * SEQ + s) * DH + c);
        dst[idx] = hv;
      }
    }
  }
}

// ---------------------------------------------------------------------------
// Cooperative staging of one 64-wide K/V block into LDS (whole workgroup).
// 64x64 halfs each = 512 16-byte chunks; 256 threads -> 2 chunks per matrix.
// ---------------------------------------------------------------------------
static __device__ inline void stage_kv(const half_t* __restrict__ Kb,
                                       const half_t* __restrict__ Vb, int kb,
                                       half_t (*kd)[72], half_t (*vd)[72], int tid) {
#pragma unroll
  for (int u = 0; u < 2; ++u) {
    const int c   = tid + u * 256;   // 0..511
    const int row = c >> 3;          // 0..63
    const int col = (c & 7) * 8;     // 0,8,..,56
    async_copy16(Kb + (size_t)(kb + row) * DH + col, &kd[row][col]);
    async_copy16(Vb + (size_t)row * SEQ + kb + col, &vd[row][col]);
  }
}

// ---------------------------------------------------------------------------
// Flash attention, no 1/sqrt(dh) scale (faithful to reference).
// One wave = 16 query rows; WG = 8 waves = 128 rows.
// K/V blocks staged once per WG into double-buffered LDS (async-to-LDS when
// available), fragments then read via ds_load_b128 with 8-wave reuse.
// Z written fp32 in [B][H][S][dh] layout (== flat reinterpret for the
// reference's raw .reshape(B,Sq,D)).
// ---------------------------------------------------------------------------
__global__ __launch_bounds__(256) void flash_attention(
    const half_t* __restrict__ Q, const half_t* __restrict__ K,
    const half_t* __restrict__ VT, float* __restrict__ Z) {
  __shared__ half_t kbuf[2][64][72];   // 18 KB
  __shared__ half_t vbuf[2][64][72];   // 18 KB
  __shared__ half_t plds[8][16][72];   // 18 KB (per-wave private P tiles)

  const int tid  = threadIdx.x;
  const int wave = tid >> 5;
  const int lane = tid & 31;
  const int n16  = lane & 15;
  const int hi   = lane >> 4;
  const int b = blockIdx.z, h = blockIdx.y;
  const int q0 = blockIdx.x * 128 + wave * 16;

  const half_t* Qb = Q  + (size_t)((b * NHEADS + h) * SEQ) * DH;
  const half_t* Kb = K  + (size_t)((b * NHEADS + h) * SEQ) * DH;
  const half_t* Vb = VT + (size_t)((b * NHEADS + h) * DH) * SEQ;

  // Kick off async staging of block 0 before loading Q fragments.
  stage_kv(Kb, Vb, 0, kbuf[0], vbuf[0], tid);

  const v16h aq0 = load_frag_a(Qb, DH, q0, 0);
  const v16h aq1 = load_frag_a(Qb, DH, q0, 32);

  float mrow[8], lrow[8];
#pragma unroll
  for (int r = 0; r < 8; ++r) { mrow[r] = -1e30f; lrow[r] = 0.0f; }
  v8f zacc[4] = {};

  int buf = 0;
  for (int kb = 0; kb < SEQ; kb += 64) {
    wait_async();       // this wave's staging chunks have landed in LDS
    __syncthreads();    // everyone's chunks visible
    if (kb + 64 < SEQ)  // overlap: stage next block while computing this one
      stage_kv(Kb, Vb, kb + 64, kbuf[buf ^ 1], vbuf[buf ^ 1], tid);

    const half_t* kl = &kbuf[buf][0][0];
    const half_t* vl = &vbuf[buf][0][0];

    // ---- S tile: 16 x 64 logits = Q(16x64) . K^T (from LDS) ----
    v8f sfr[4];
#pragma unroll
    for (int j = 0; j < 4; ++j) {
      v16h bk0 = load_frag_bt(kl, 72, j * 16, 0);
      v16h bk1 = load_frag_bt(kl, 72, j * 16, 32);
      v8f z0 = {};
      z0 = __builtin_amdgcn_wmma_f32_16x16x32_f16(false, aq0, false, bk0, (short)0, z0, false, false);
      z0 = __builtin_amdgcn_wmma_f32_16x16x32_f16(false, aq1, false, bk1, (short)0, z0, false, false);
      sfr[j] = z0;
    }
    // ---- online softmax (rows live in VGPR index + lane-group halves) ----
    float nm[8];
#pragma unroll
    for (int r = 0; r < 8; ++r) {
      float v = fmaxf(fmaxf(sfr[0][r], sfr[1][r]), fmaxf(sfr[2][r], sfr[3][r]));
#pragma unroll
      for (int d = 1; d < 16; d <<= 1) v = fmaxf(v, __shfl_xor(v, d, 32));
      nm[r] = fmaxf(mrow[r], v);
    }
#pragma unroll
    for (int r = 0; r < 8; ++r) {
      const float sc = __expf(mrow[r] - nm[r]);
      lrow[r] *= sc;
#pragma unroll
      for (int j = 0; j < 4; ++j) zacc[j][r] *= sc;
      float rs = 0.0f;
#pragma unroll
      for (int j = 0; j < 4; ++j) {
        const float p = __expf(sfr[j][r] - nm[r]);
        sfr[j][r] = p;
        rs += p;
      }
#pragma unroll
      for (int d = 1; d < 16; d <<= 1) rs += __shfl_xor(rs, d, 32);
      lrow[r] += rs;
      mrow[r] = nm[r];
    }
    // ---- relayout P (C-layout -> A-layout) through per-wave LDS slab ----
#pragma unroll
    for (int j = 0; j < 4; ++j)
#pragma unroll
      for (int r = 0; r < 8; ++r)
        plds[wave][hi * 8 + r][j * 16 + n16] = (half_t)sfr[j][r];
    asm volatile("s_wait_dscnt 0x0" ::: "memory");
    const v16h pa0 = load_frag_a(&plds[wave][0][0], 72, 0, 0);
    const v16h pa1 = load_frag_a(&plds[wave][0][0], 72, 0, 32);
    // ---- Z += P(16x64) . V(64x64) (V block from LDS, transposed layout) ----
#pragma unroll
    for (int j = 0; j < 4; ++j) {
      v16h bv0 = load_frag_bt(vl, 72, j * 16, 0);
      v16h bv1 = load_frag_bt(vl, 72, j * 16, 32);
      v8f z = zacc[j];
      z = __builtin_amdgcn_wmma_f32_16x16x32_f16(false, pa0, false, bv0, (short)0, z, false, false);
      z = __builtin_amdgcn_wmma_f32_16x16x32_f16(false, pa1, false, bv1, (short)0, z, false, false);
      zacc[j] = z;
    }
    __syncthreads();    // all waves done reading buf before it is re-staged
    buf ^= 1;
  }

  float inv[8];
#pragma unroll
  for (int r = 0; r < 8; ++r) inv[r] = 1.0f / lrow[r];
  float* Zb = Z + (size_t)((b * NHEADS + h) * SEQ + q0) * DH;
#pragma unroll
  for (int j = 0; j < 4; ++j)
#pragma unroll
    for (int r = 0; r < 8; ++r)
      Zb[(size_t)(hi * 8 + r) * DH + j * 16 + n16] = zacc[j][r] * inv[r];
}

// ---------------------------------------------------------------------------
// Fused residual + LayerNorm: one block (256 threads) per row of 1024.
// ---------------------------------------------------------------------------
__global__ __launch_bounds__(256) void add_layernorm(
    const float* __restrict__ Zflat, const float* __restrict__ dec,
    const float* __restrict__ gamma, const float* __restrict__ beta,
    float* __restrict__ out) {
  __shared__ float red[16];
  const int row = blockIdx.x;
  const int t = threadIdx.x;
  const float4 z4 = ((const float4*)(Zflat + (size_t)row * DMODEL))[t];
  const float4 d4 = ((const float4*)(dec   + (size_t)row * DMODEL))[t];
  float4 x;
  x.x = z4.x + d4.x; x.y = z4.y + d4.y; x.z = z4.z + d4.z; x.w = z4.w + d4.w;

  float s  = x.x + x.y + x.z + x.w;
  float sq = fmaf(x.x, x.x, fmaf(x.y, x.y, fmaf(x.z, x.z, x.w * x.w)));
#pragma unroll
  for (int d = 1; d < 32; d <<= 1) {
    s  += __shfl_xor(s,  d, 32);
    sq += __shfl_xor(sq, d, 32);
  }
  const int wave = t >> 5, lane = t & 31;
  if (lane == 0) { red[wave] = s; red[8 + wave] = sq; }
  __syncthreads();
  float ts = 0.0f, tq = 0.0f;
#pragma unroll
  for (int i = 0; i < 8; ++i) { ts += red[i]; tq += red[8 + i]; }

  const float mu  = ts * (1.0f / DMODEL);
  const float var = tq * (1.0f / DMODEL) - mu * mu;
  const float rs  = rsqrtf(var + LN_EPS);
  const float4 g4 = ((const float4*)gamma)[t];
  const float4 b4 = ((const float4*)beta)[t];
  float4 y;
  y.x = (x.x - mu) * rs * g4.x + b4.x;
  y.y = (x.y - mu) * rs * g4.y + b4.y;
  y.z = (x.z - mu) * rs * g4.z + b4.z;
  y.w = (x.w - mu) * rs * g4.w + b4.w;
  ((float4*)(out + (size_t)row * DMODEL))[t] = y;
}

// ---------------------------------------------------------------------------
// Host-side launch. Workspace layout (~62 MB total):
//   decH, encH   : 4096x1024 f16 each
//   WqH,WkH,WvH  : 1024x1024 f16 each
//   Qh, Kh       : [B][H][S][dh] f16
//   Vth          : [B][H][dh][S] f16 (transposed for B-fragment loads)
//   Zf           : [B][H][S][dh] f32 (flat == reference reshape)
// ---------------------------------------------------------------------------
extern "C" void kernel_launch(void* const* d_in, const int* in_sizes, int n_in,
                              void* d_out, int out_size, void* d_ws, size_t ws_size,
                              hipStream_t stream) {
  (void)in_sizes; (void)n_in; (void)out_size; (void)ws_size;
  const float* encoded = (const float*)d_in[0];
  const float* decoded = (const float*)d_in[1];
  const float* Wq = (const float*)d_in[2];
  const float* bq = (const float*)d_in[3];
  const float* Wk = (const float*)d_in[4];
  const float* bk = (const float*)d_in[5];
  const float* Wv = (const float*)d_in[6];
  const float* bv = (const float*)d_in[7];
  const float* gamma = (const float*)d_in[8];
  const float* beta  = (const float*)d_in[9];
  float* out = (float*)d_out;

  const size_t ACT = (size_t)MROWS * DMODEL;   // 4,194,304 elems
  const size_t WSZ = (size_t)DMODEL * DMODEL;  // 1,048,576 elems
  half_t* decH = (half_t*)d_ws;
  half_t* encH = decH + ACT;
  half_t* WqH  = encH + ACT;
  half_t* WkH  = WqH + WSZ;
  half_t* WvH  = WkH + WSZ;
  half_t* Qh   = WvH + WSZ;
  half_t* Kh   = Qh + ACT;
  half_t* Vth  = Kh + ACT;
  float*  Zf   = (float*)(Vth + ACT);

  // 1) fp32 -> fp16
  cvt_f32_to_f16<<<(int)(ACT / 4 / 256), 256, 0, stream>>>(decoded, decH, (int)(ACT / 4));
  cvt_f32_to_f16<<<(int)(ACT / 4 / 256), 256, 0, stream>>>(encoded, encH, (int)(ACT / 4));
  cvt_f32_to_f16<<<(int)(WSZ / 4 / 256), 256, 0, stream>>>(Wq, WqH, (int)(WSZ / 4));
  cvt_f32_to_f16<<<(int)(WSZ / 4 / 256), 256, 0, stream>>>(Wk, WkH, (int)(WSZ / 4));
  cvt_f32_to_f16<<<(int)(WSZ / 4 / 256), 256, 0, stream>>>(Wv, WvH, (int)(WSZ / 4));

  // 2) QKV projections (WMMA)
  dim3 pgrid(MROWS / 128, DMODEL / 128);
  proj_gemm<<<pgrid, 256, 0, stream>>>(decH, WqH, bq, Qh, 0);
  proj_gemm<<<pgrid, 256, 0, stream>>>(encH, WkH, bk, Kh, 0);
  proj_gemm<<<pgrid, 256, 0, stream>>>(encH, WvH, bv, Vth, 1);

  // 3) attention (WMMA flash, async LDS staging)
  flash_attention<<<dim3(SEQ / 128, NHEADS, BATCH), 256, 0, stream>>>(Qh, Kh, Vth, Zf);

  // 4) residual + layernorm
  add_layernorm<<<MROWS, 256, 0, stream>>>(Zf, decoded, gamma, beta, out);
}
